// SpectralMessage_56659208569405
// MI455X (gfx1250) — matrix operand order repaired and checked by reading, not compiled, not run on hardware
//
#include <hip/hip_runtime.h>

typedef __attribute__((ext_vector_type(16))) _Float16 v16h;
typedef __attribute__((ext_vector_type(8)))  _Float16 v8h;
typedef __attribute__((ext_vector_type(4)))  _Float16 v4h;
typedef __attribute__((ext_vector_type(8)))  float    v8f;
typedef __attribute__((ext_vector_type(4)))  float    v4f;

#define LDSTRIDE 40  // halves per LDS row: 80B rows -> 16B-aligned frags, odd bank stride

// ---------------------------------------------------------------------------
// Tiled WMMA GEMM: C[M,N] = act(A[M,K] @ B[K,N]) (+R)   (f32 in/out, f16 MACs)
// 256 threads = 8 waves (4x2). Block tile 128x64, BK=32. Wave tile 32x32 =
// 4 x v_wmma_f32_16x16x32_f16 per K-step. Software-pipelined global loads.
// All global offsets kept in 32-bit with loop-invariant bases so the K-loop
// induction is a single scalar add (no v_mul_u64 chains).
// ---------------------------------------------------------------------------
template <bool RELU, bool RESID>
__global__ __launch_bounds__(256) void gemm16(const float* __restrict__ A,
                                              const float* __restrict__ B,
                                              const float* __restrict__ R,
                                              float* __restrict__ C,
                                              int M, int N, int K) {
  __shared__ __align__(16) _Float16 sA[128 * LDSTRIDE];  // [m][k]
  __shared__ __align__(16) _Float16 sB[64 * LDSTRIDE];   // [n][k] (transposed)

  const int t    = threadIdx.x;
  const int wave = t >> 5;
  const int lane = t & 31;
  const int l16  = lane & 15;
  const int hi   = lane >> 4;
  const int waveM = wave >> 1;  // 0..3 -> 32-row strip
  const int waveN = wave & 1;   // 0..1 -> 32-col strip
  const int m0 = blockIdx.y * 128;
  const int n0 = blockIdx.x * 64;
  const int bn = t & 63;        // B-tile column owned by this thread

  v8f acc[2][2] = {{{}, {}}, {{}, {}}};

  // ---- loop-invariant 32-bit offset bases ------------------------------
  int abase[4];   // A[(m0+r)*K + c4*4], r/c4 fixed per slot
#pragma unroll
  for (int i = 0; i < 4; ++i) {
    int s = i * 256 + t;  // float4 slot in 128x32 tile
    int r = s >> 3, c4 = s & 7;
    abase[i] = (m0 + r) * K + c4 * 4;
  }
  int bbase[2];   // B[(kg*4)*N + n0 + bn], kg fixed per slot
#pragma unroll
  for (int i = 0; i < 2; ++i) bbase[i] = (((t >> 6) + i * 4) * 4) * N + n0 + bn;

  // ---- pipelined register staging -------------------------------------
  v4f ra[4];        // A tile: 128x32 f32 = 1024 float4, 4 per thread
  float rb[2][4];   // B tile: 32x64 f32, 2 column-segments of 4 per thread

  auto load_tiles = [&](int kk) {
    const int koffB = kk * N;  // scalar (SALU) per iteration
#pragma unroll
    for (int i = 0; i < 4; ++i) ra[i] = *(const v4f*)&A[abase[i] + kk];
#pragma unroll
    for (int i = 0; i < 2; ++i) {
      const float* p = B + bbase[i] + koffB;
#pragma unroll
      for (int q = 0; q < 4; ++q) rb[i][q] = p[q * N];  // q*N hoisted
    }
  };

  auto store_tiles = [&]() {
#pragma unroll
    for (int i = 0; i < 4; ++i) {
      int s = i * 256 + t;
      int r = s >> 3, c4 = s & 7;
      v4h h;
#pragma unroll
      for (int q = 0; q < 4; ++q) h[q] = (_Float16)ra[i][q];
      *(v4h*)&sA[r * LDSTRIDE + c4 * 4] = h;
    }
#pragma unroll
    for (int i = 0; i < 2; ++i) {
      int kg = (t >> 6) + i * 4;
      v4h h;
#pragma unroll
      for (int q = 0; q < 4; ++q) h[q] = (_Float16)rb[i][q];
      *(v4h*)&sB[bn * LDSTRIDE + kg * 4] = h;
    }
  };

  load_tiles(0);

  for (int kk = 0; kk < K; kk += 32) {
    store_tiles();
    __syncthreads();

    if (kk + 32 < K) load_tiles(kk + 32);  // overlap with WMMA below

    // A fragments: row m; K halves {hi*8..+7, 16+hi*8..+7}
    v16h af[2];
#pragma unroll
    for (int mi = 0; mi < 2; ++mi) {
      int am = waveM * 32 + mi * 16 + l16;
      v8h lo = *(const v8h*)&sA[am * LDSTRIDE + hi * 8];
      v8h hh = *(const v8h*)&sA[am * LDSTRIDE + 16 + hi * 8];
#pragma unroll
      for (int e = 0; e < 8; ++e) { af[mi][e] = lo[e]; af[mi][e + 8] = hh[e]; }
    }
    // B fragments: col n; K halves {hi*8..+7, 16+hi*8..+7}
    v16h bf[2];
#pragma unroll
    for (int ni = 0; ni < 2; ++ni) {
      int bc = waveN * 32 + ni * 16 + l16;
      v8h lo = *(const v8h*)&sB[bc * LDSTRIDE + hi * 8];
      v8h hh = *(const v8h*)&sB[bc * LDSTRIDE + 16 + hi * 8];
#pragma unroll
      for (int e = 0; e < 8; ++e) { bf[ni][e] = lo[e]; bf[ni][e + 8] = hh[e]; }
    }

#pragma unroll
    for (int mi = 0; mi < 2; ++mi)
#pragma unroll
      for (int ni = 0; ni < 2; ++ni)
        acc[mi][ni] = __builtin_amdgcn_wmma_f32_16x16x32_f16(
            false, af[mi], false, bf[ni], (short)0, acc[mi][ni], false, false);

    __syncthreads();
  }

  // Epilogue: C/D layout -> M = hi*8 + r, N = l16
#pragma unroll
  for (int mi = 0; mi < 2; ++mi) {
#pragma unroll
    for (int ni = 0; ni < 2; ++ni) {
#pragma unroll
      for (int r = 0; r < 8; ++r) {
        int row = m0 + waveM * 32 + mi * 16 + hi * 8 + r;
        int col = n0 + waveN * 32 + ni * 16 + l16;
        float v = acc[mi][ni][r];
        if (RELU) v = fmaxf(v, 0.f);
        size_t idx = (size_t)row * N + col;
        if (RESID) v += R[idx];
        C[idx] = v;
      }
    }
  }
}

// ---------------------------------------------------------------------------
// conv1: gather freq_table rows per 3x3 neighborhood, 51 -> 10 channels.
// One block = one batch slice of 256 pixels (25 blocks per batch).
// ---------------------------------------------------------------------------
#define NCLS 151
__global__ __launch_bounds__(256) void conv1_k(const int* __restrict__ obj,
                                               const float* __restrict__ freq,
                                               const float* __restrict__ w1,
                                               float* __restrict__ x1) {
  __shared__ float sw[4590];  // [oc][c][3][3]
  __shared__ int sobj[80];
  const int b   = blockIdx.x / 25;
  const int pix = (blockIdx.x % 25) * 256 + threadIdx.x;
  for (int w = threadIdx.x; w < 4590; w += 256) sw[w] = w1[w];
  if (threadIdx.x < 80) sobj[threadIdx.x] = obj[b * 80 + threadIdx.x];
  __syncthreads();
  const int i = pix / 80, j = pix % 80;
  float acc[10];
#pragma unroll
  for (int oc = 0; oc < 10; ++oc) acc[oc] = 0.f;
  for (int di = 0; di < 3; ++di) {
    int ii = i + di - 1;
    if (ii < 0 || ii >= 80) continue;
    for (int dj = 0; dj < 3; ++dj) {
      int jj = j + dj - 1;
      if (jj < 0 || jj >= 80) continue;
      int dd = di * 3 + dj;
      const float* f = freq + ((size_t)sobj[ii] * NCLS + sobj[jj]) * 51;
      for (int c = 0; c < 51; ++c) {
        float x = f[c];
#pragma unroll
        for (int oc = 0; oc < 10; ++oc)
          acc[oc] = fmaf(x, sw[oc * 459 + c * 9 + dd], acc[oc]);
      }
    }
  }
  for (int oc = 0; oc < 10; ++oc)
    x1[(((size_t)b * 10 + oc) * 80 + i) * 80 + j] = acc[oc];
}

// conv2(3x3, 10->5) + conv3(1x1, 5->1) + sigmoid -> adj[B,80,80]
__global__ __launch_bounds__(256) void conv23_k(const float* __restrict__ x1,
                                                const float* __restrict__ w2,
                                                const float* __restrict__ w3,
                                                float* __restrict__ adj) {
  __shared__ float sw2[450];
  __shared__ float sw3[5];
  const int b   = blockIdx.x / 25;
  const int pix = (blockIdx.x % 25) * 256 + threadIdx.x;
  for (int w = threadIdx.x; w < 450; w += 256) sw2[w] = w2[w];
  if (threadIdx.x < 5) sw3[threadIdx.x] = w3[threadIdx.x];
  __syncthreads();
  const int i = pix / 80, j = pix % 80;
  float acc[5] = {0.f, 0.f, 0.f, 0.f, 0.f};
  for (int di = 0; di < 3; ++di) {
    int ii = i + di - 1;
    if (ii < 0 || ii >= 80) continue;
    for (int dj = 0; dj < 3; ++dj) {
      int jj = j + dj - 1;
      if (jj < 0 || jj >= 80) continue;
      int dd = di * 3 + dj;
      for (int ic = 0; ic < 10; ++ic) {
        float x = x1[(((size_t)b * 10 + ic) * 80 + ii) * 80 + jj];
#pragma unroll
        for (int oc = 0; oc < 5; ++oc)
          acc[oc] = fmaf(x, sw2[oc * 90 + ic * 9 + dd], acc[oc]);
      }
    }
  }
  float z = 0.f;
#pragma unroll
  for (int oc = 0; oc < 5; ++oc) z = fmaf(acc[oc], sw3[oc], z);
  adj[(size_t)b * 6400 + pix] = 1.f / (1.f + __expf(-z));
}

// d[b,i] = (sum_j (adj+I)[i,j] + 1e-5)^-1/2
__global__ void deg_k(const float* __restrict__ adj, float* __restrict__ dvec) {
  int t = blockIdx.x * blockDim.x + threadIdx.x;
  if (t >= 64 * 80) return;
  int b = t / 80, i = t % 80;
  const float* row = adj + (size_t)b * 6400 + i * 80;
  float s = 1.0f;  // diagonal of A_hat
  for (int j = 0; j < 80; ++j) s += row[j];
  dvec[t] = rsqrtf(s + 1e-5f);
}

// L_hat = d_i * (adj + I) * d_j
__global__ void lhat_k(const float* __restrict__ adj,
                       const float* __restrict__ dvec,
                       float* __restrict__ lhat) {
  int t = blockIdx.x * blockDim.x + threadIdx.x;
  if (t >= 64 * 6400) return;
  int b = t / 6400, r = t % 6400, i = r / 80, j = r % 80;
  float a = adj[t] + ((i == j) ? 1.f : 0.f);
  lhat[t] = dvec[b * 80 + i] * a * dvec[b * 80 + j];
}

// ofl_u[b,n,:] = L_hat[b,n,:] @ obj_u1[b,:,:]   (one block per (b,n))
__global__ __launch_bounds__(256) void oflu_k(const float* __restrict__ lhat,
                                              const float* __restrict__ u1,
                                              float* __restrict__ out) {
  __shared__ float lrow[80];
  const int b = blockIdx.x / 80, n = blockIdx.x % 80;
  if (threadIdx.x < 80)
    lrow[threadIdx.x] = lhat[(size_t)b * 6400 + n * 80 + threadIdx.x];
  __syncthreads();
  const float* ub = u1 + (size_t)b * 80 * 1024;
  for (int k = threadIdx.x; k < 1024; k += 256) {
    float s = 0.f;
    for (int m = 0; m < 80; ++m) s = fmaf(lrow[m], ub[(size_t)m * 1024 + k], s);
    out[((size_t)b * 80 + n) * 1024 + k] = s;
  }
}

// ---------------------------------------------------------------------------
extern "C" void kernel_launch(void* const* d_in, const int* in_sizes, int n_in,
                              void* d_out, int out_size, void* d_ws,
                              size_t ws_size, hipStream_t stream) {
  const float* E    = (const float*)d_in[0];  // [64,80,4096]
  const int*   obj  = (const int*)d_in[1];    // [64,80]
  const float* freq = (const float*)d_in[2];  // [151*151,51]
  const float* Wc   = (const float*)d_in[3];  // [4096,1024]
  const float* Wou1 = (const float*)d_in[4];  // [1024,1024]
  const float* Wofc = (const float*)d_in[5];  // [1024,1024]
  const float* Wd   = (const float*)d_in[6];  // [1024,4096]
  const float* c1   = (const float*)d_in[7];  // [10,51,3,3]
  const float* c2   = (const float*)d_in[8];  // [5,10,3,3]
  const float* c3   = (const float*)d_in[9];  // [1,5,1,1]
  float* out = (float*)d_out;                 // [64,80,4096]

  const size_t SZ = (size_t)5120 * 1024;      // elements per big buffer
  float* bufA = (float*)d_ws;                 // comp, later ofl_u
  float* bufB = bufA + SZ;                    // obj_u1
  float* bufC = bufB + SZ;                    // x1 (16.4MB < 21MB), later u_m
  float* adj  = bufC + SZ;                    // [64,80,80]
  float* lhat = adj + (size_t)64 * 6400;      // [64,80,80]
  float* dvec = lhat + (size_t)64 * 6400;     // [64,80]

  dim3 blk(256);

  // comp = ReLU(E @ W_comp)            [5120,4096] x [4096,1024]
  gemm16<true, false><<<dim3(16, 40), blk, 0, stream>>>(E, Wc, nullptr, bufA,
                                                        5120, 1024, 4096);
  // obj_u1 = ReLU(comp @ W_ou1)        [5120,1024] x [1024,1024]
  gemm16<true, false><<<dim3(16, 40), blk, 0, stream>>>(bufA, Wou1, nullptr,
                                                        bufB, 5120, 1024, 1024);
  // adjacency path
  conv1_k<<<64 * 25, blk, 0, stream>>>(obj, freq, c1, bufC);
  conv23_k<<<64 * 25, blk, 0, stream>>>(bufC, c2, c3, adj);
  deg_k<<<20, blk, 0, stream>>>(adj, dvec);
  lhat_k<<<1600, blk, 0, stream>>>(adj, dvec, lhat);
  // ofl_u = L_hat @ obj_u1  -> bufA (comp dead)
  oflu_k<<<64 * 80, blk, 0, stream>>>(lhat, bufB, bufA);
  // u_m = ReLU(ofl_u @ W_ofc) -> bufC (x1 dead)
  gemm16<true, false><<<dim3(16, 40), blk, 0, stream>>>(bufA, Wofc, nullptr,
                                                        bufC, 5120, 1024, 1024);
  // out = ReLU(u_m @ W_decomp) + E     [5120,1024] x [1024,4096]
  gemm16<true, true><<<dim3(64, 40), blk, 0, stream>>>(bufC, Wd, E, out, 5120,
                                                       4096, 1024);
}
